// DirectionCrossAttention_17549236371977
// MI455X (gfx1250) — matrix operand order repaired and checked by reading, not compiled, not run on hardware
//
#include <hip/hip_runtime.h>
#include <hip/hip_bf16.h>
#include <math.h>

typedef _Float16 f16x16 __attribute__((ext_vector_type(16)));
typedef _Float16 f16x8  __attribute__((ext_vector_type(8)));
typedef float    f32x8  __attribute__((ext_vector_type(8)));

#define B_   2
#define N_   6
#define D_   128
#define Q_   2500
#define QP_  2512     /* 157*16 */
#define QT_  157
#define HK_  420
#define K_   2520
#define KP_  2528     /* 79*32 */
#define KT_  158
#define M_   4
#define DH_  32
#define HD_  128
#define SCALE_ 0.17677669529663687f   /* 32^-0.5 */

union U16 { f16x16 v; f16x8 c[2]; };

__device__ __forceinline__ f32x8 wmma_f16(f16x16 a, f16x16 b, f32x8 c) {
  return __builtin_amdgcn_wmma_f32_16x16x32_f16(false, a, false, b, (short)0, c, false, false);
}

__device__ __forceinline__ float redSum32(float v) {
#pragma unroll
  for (int o = 16; o >= 1; o >>= 1) v += __shfl_xor(v, o, 32);
  return v;
}
__device__ __forceinline__ float redSum16(float v) {
  v += __shfl_xor(v, 1, 32); v += __shfl_xor(v, 2, 32);
  v += __shfl_xor(v, 4, 32); v += __shfl_xor(v, 8, 32);
  return v;
}
__device__ __forceinline__ float redMax16(float v) {
  v = fmaxf(v, __shfl_xor(v, 1, 32)); v = fmaxf(v, __shfl_xor(v, 2, 32));
  v = fmaxf(v, __shfl_xor(v, 4, 32)); v = fmaxf(v, __shfl_xor(v, 8, 32));
  return v;
}

/* ---- Kernel 1: transpose + f16-convert projection weights (col-major, k contiguous) ---- */
__global__ void k_convert_w(const float* __restrict__ wq, const float* __restrict__ wk,
                            const float* __restrict__ wv,
                            _Float16* __restrict__ wqT, _Float16* __restrict__ wkT,
                            _Float16* __restrict__ wvT) {
  int i = blockIdx.x * blockDim.x + threadIdx.x;
  if (i >= D_ * HD_) return;
  int k = i / HD_, n = i % HD_;
  wqT[n * D_ + k] = (_Float16)wq[i];
  wkT[n * D_ + k] = (_Float16)wk[i];
  wvT[n * D_ + k] = (_Float16)wv[i];
}

/* ---- Kernel 2: Q projection: qsum = (LN(x0)+LN(x1)) @ wq + 2*bq, f16 out ---- */
__global__ void k_proj_q(const float* __restrict__ qin,
                         const float* __restrict__ lnq_g, const float* __restrict__ lnq_b,
                         const _Float16* __restrict__ wqT, const float* __restrict__ bq,
                         _Float16* __restrict__ qsum) {
  __shared__ __align__(32) _Float16 sA[4][16][D_];
  const int w = threadIdx.x >> 5, lane = threadIdx.x & 31;
  const int tile = blockIdx.x * 4 + w;
  if (tile >= B_ * QT_) return;
  const int b = tile / QT_, qt = tile % QT_;
  const int hi = lane >> 4, ln = lane & 15;

  for (int r = 0; r < 16; ++r) {
    const int qi = qt * 16 + r;
    if (qi < Q_) {
      float v0[4], v1[4];
      float s0 = 0.f, ss0 = 0.f, s1 = 0.f, ss1 = 0.f;
#pragma unroll
      for (int t = 0; t < 4; ++t) {
        int d = lane * 4 + t;
        float a = qin[((size_t)((b * N_ + 0) * D_ + d)) * Q_ + qi];
        float c = qin[((size_t)((b * N_ + 1) * D_ + d)) * Q_ + qi];
        v0[t] = a; v1[t] = c;
        s0 += a; ss0 += a * a; s1 += c; ss1 += c * c;
      }
      s0 = redSum32(s0); ss0 = redSum32(ss0); s1 = redSum32(s1); ss1 = redSum32(ss1);
      float m0 = s0 * (1.f / 128.f), m1 = s1 * (1.f / 128.f);
      float r0 = rsqrtf(ss0 * (1.f / 128.f) - m0 * m0 + 1e-5f);
      float r1 = rsqrtf(ss1 * (1.f / 128.f) - m1 * m1 + 1e-5f);
#pragma unroll
      for (int t = 0; t < 4; ++t) {
        int d = lane * 4 + t;
        float g = lnq_g[d], be = lnq_b[d];
        float val = ((v0[t] - m0) * r0) * g + be + ((v1[t] - m1) * r1) * g + be;
        sA[w][r][d] = (_Float16)val;
      }
    } else {
#pragma unroll
      for (int t = 0; t < 4; ++t) sA[w][r][lane * 4 + t] = (_Float16)0.f;
    }
  }
  asm volatile("s_wait_dscnt 0" ::: "memory");

  for (int nt = 0; nt < 8; ++nt) {
    const int col = nt * 16 + ln;
    f32x8 acc;
#pragma unroll
    for (int i = 0; i < 8; ++i) acc[i] = 0.f;
#pragma unroll
    for (int ks = 0; ks < 4; ++ks) {
      U16 ua;
      ua.c[0] = *(const f16x8*)&sA[w][ln][ks * 32 + 16 * hi];
      ua.c[1] = *(const f16x8*)&sA[w][ln][ks * 32 + 16 * hi + 8];
      f16x16 vb = *(const f16x16*)&wqT[col * D_ + ks * 32 + 16 * hi];
      acc = wmma_f16(ua.v, vb, acc);
    }
    const float bias = 2.f * bq[col];
#pragma unroll
    for (int r = 0; r < 8; ++r) {
      int qrow = qt * 16 + r + 8 * hi;
      qsum[((size_t)(b * QP_ + qrow)) * HD_ + col] = (_Float16)(acc[r] + bias);
    }
  }
}

/* ---- Kernel 3: K/V projections, kp key-major (dh contig), vp transposed (key contig) ---- */
__global__ void k_proj_kv(const float* __restrict__ kin, const float* __restrict__ vin,
                          const float* __restrict__ lnk_g, const float* __restrict__ lnk_b,
                          const float* __restrict__ lnv_g, const float* __restrict__ lnv_b,
                          const _Float16* __restrict__ wkT, const float* __restrict__ bk,
                          const _Float16* __restrict__ wvT, const float* __restrict__ bv,
                          _Float16* __restrict__ kp, _Float16* __restrict__ vpt) {
  __shared__ __align__(32) _Float16 sK[4][16][D_];
  __shared__ __align__(32) _Float16 sV[4][16][D_];
  const int w = threadIdx.x >> 5, lane = threadIdx.x & 31;
  const int tile = blockIdx.x * 4 + w;
  if (tile >= B_ * KT_) return;
  const int b = tile / KT_, kt = tile % KT_;
  const int hi = lane >> 4, ln = lane & 15;

  for (int r = 0; r < 16; ++r) {
    const int kidx = kt * 16 + r;
    if (kidx < K_) {
      const int n = kidx / HK_, off = kidx % HK_;
      float xk[4], xv[4];
      float sk = 0.f, ssk = 0.f, sv = 0.f, ssv = 0.f;
#pragma unroll
      for (int t = 0; t < 4; ++t) {
        int d = lane * 4 + t;
        float a = kin[((size_t)((b * N_ + n) * D_ + d)) * HK_ + off];
        float c = vin[((size_t)((b * N_ + n) * D_ + d)) * HK_ + off];
        xk[t] = a; xv[t] = c;
        sk += a; ssk += a * a; sv += c; ssv += c * c;
      }
      sk = redSum32(sk); ssk = redSum32(ssk); sv = redSum32(sv); ssv = redSum32(ssv);
      float mk = sk * (1.f / 128.f), mv = sv * (1.f / 128.f);
      float rk = rsqrtf(ssk * (1.f / 128.f) - mk * mk + 1e-5f);
      float rv = rsqrtf(ssv * (1.f / 128.f) - mv * mv + 1e-5f);
#pragma unroll
      for (int t = 0; t < 4; ++t) {
        int d = lane * 4 + t;
        sK[w][r][d] = (_Float16)(((xk[t] - mk) * rk) * lnk_g[d] + lnk_b[d]);
        sV[w][r][d] = (_Float16)(((xv[t] - mv) * rv) * lnv_g[d] + lnv_b[d]);
      }
    } else {
#pragma unroll
      for (int t = 0; t < 4; ++t) {
        sK[w][r][lane * 4 + t] = (_Float16)0.f;
        sV[w][r][lane * 4 + t] = (_Float16)0.f;
      }
    }
  }
  asm volatile("s_wait_dscnt 0" ::: "memory");

  for (int nt = 0; nt < 8; ++nt) {
    const int col = nt * 16 + ln;
    f32x8 accK, accV;
#pragma unroll
    for (int i = 0; i < 8; ++i) { accK[i] = 0.f; accV[i] = 0.f; }
#pragma unroll
    for (int ks = 0; ks < 4; ++ks) {
      U16 ua, uv;
      ua.c[0] = *(const f16x8*)&sK[w][ln][ks * 32 + 16 * hi];
      ua.c[1] = *(const f16x8*)&sK[w][ln][ks * 32 + 16 * hi + 8];
      f16x16 vbk = *(const f16x16*)&wkT[col * D_ + ks * 32 + 16 * hi];
      accK = wmma_f16(ua.v, vbk, accK);
      uv.c[0] = *(const f16x8*)&sV[w][ln][ks * 32 + 16 * hi];
      uv.c[1] = *(const f16x8*)&sV[w][ln][ks * 32 + 16 * hi + 8];
      f16x16 vbv = *(const f16x16*)&wvT[col * D_ + ks * 32 + 16 * hi];
      accV = wmma_f16(uv.v, vbv, accV);
    }
    const int m = col >> 5, dh = col & 31;
    const float bK = bk[col], bV = bv[col];
#pragma unroll
    for (int r = 0; r < 8; ++r) {
      int key = kt * 16 + r + 8 * hi;
      kp[((size_t)((b * M_ + m) * KP_ + key)) * DH_ + dh] = (_Float16)(accK[r] + bK);
      vpt[((size_t)((b * M_ + m) * DH_ + dh)) * KP_ + key] = (_Float16)(accV[r] + bV);
    }
  }
}

/* ---- Kernel 4: flash attention, one wave per (b, m, 16-query tile) ---- */
__global__ void k_attn(const _Float16* __restrict__ qsum,
                       const _Float16* __restrict__ kp,
                       const _Float16* __restrict__ vpt,
                       float* __restrict__ att) {
  __shared__ __align__(32) _Float16 sP[8][16][32];
  const int w = threadIdx.x >> 5, lane = threadIdx.x & 31;
  const int task = blockIdx.x * 8 + w;
  if (task >= B_ * M_ * QT_) return;
  const int bm = task / QT_, qt = task % QT_;
  const int b = bm >> 2, m = bm & 3;
  const int hi = lane >> 4, ln = lane & 15;

  U16 uq;
  {
    const _Float16* base = &qsum[((size_t)(b * QP_ + qt * 16 + ln)) * HD_ + m * DH_];
    uq.c[0] = *(const f16x8*)(base + 16 * hi);
    uq.c[1] = *(const f16x8*)(base + 16 * hi + 8);
  }

  f32x8 acc0, acc1;
  float mrun[8], lrun[8];
#pragma unroll
  for (int r = 0; r < 8; ++r) { acc0[r] = 0.f; acc1[r] = 0.f; mrun[r] = -3.0e38f; lrun[r] = 0.f; }

  for (int kb = 0; kb < KP_ / 32; ++kb) {
    const int key0 = kb * 32;
    f16x16 bk0 = *(const f16x16*)&kp[((size_t)(bm * KP_ + key0 + ln)) * DH_ + 16 * hi];
    f16x16 bk1 = *(const f16x16*)&kp[((size_t)(bm * KP_ + key0 + 16 + ln)) * DH_ + 16 * hi];
    f32x8 zc;
#pragma unroll
    for (int i = 0; i < 8; ++i) zc[i] = 0.f;
    f32x8 s0 = wmma_f16(uq.v, bk0, zc);
    f32x8 s1 = wmma_f16(uq.v, bk1, zc);
    const bool ok0 = (key0 + ln) < K_;
    const bool ok1 = (key0 + 16 + ln) < K_;
#pragma unroll
    for (int r = 0; r < 8; ++r) {
      float a0 = ok0 ? s0[r] * SCALE_ : -1.0e30f;
      float a1 = ok1 ? s1[r] * SCALE_ : -1.0e30f;
      float bmax = redMax16(fmaxf(a0, a1));
      float mn = fmaxf(mrun[r], bmax);
      float f = __expf(mrun[r] - mn);
      float p0 = __expf(a0 - mn);
      float p1 = __expf(a1 - mn);
      float rs = redSum16(p0 + p1);
      lrun[r] = lrun[r] * f + rs;
      mrun[r] = mn;
      acc0[r] *= f; acc1[r] *= f;
      sP[w][r + 8 * hi][ln] = (_Float16)p0;
      sP[w][r + 8 * hi][16 + ln] = (_Float16)p1;
    }
    asm volatile("s_wait_dscnt 0" ::: "memory");
    U16 up;
    up.c[0] = *(const f16x8*)&sP[w][ln][16 * hi];
    up.c[1] = *(const f16x8*)&sP[w][ln][16 * hi + 8];
    f16x16 bv0 = *(const f16x16*)&vpt[((size_t)(bm * DH_ + ln)) * KP_ + key0 + 16 * hi];
    f16x16 bv1 = *(const f16x16*)&vpt[((size_t)(bm * DH_ + 16 + ln)) * KP_ + key0 + 16 * hi];
    acc0 = wmma_f16(up.v, bv0, acc0);
    acc1 = wmma_f16(up.v, bv1, acc1);
    asm volatile("" ::: "memory");
  }
#pragma unroll
  for (int r = 0; r < 8; ++r) {
    float inv = 1.f / lrun[r];
    int qrow = qt * 16 + r + 8 * hi;
    float* o = &att[((size_t)(b * QP_ + qrow)) * HD_ + m * DH_];
    o[ln] = acc0[r] * inv;
    o[16 + ln] = acc1[r] * inv;
  }
}

/* ---- Kernel 5: proj + skip + LN + MLP(GELU) + LN, one wave per (b,q) row ---- */
__global__ void k_epilogue(const float* __restrict__ att, const float* __restrict__ skip,
                           const float* __restrict__ wp, const float* __restrict__ bp,
                           const float* __restrict__ ln1_g, const float* __restrict__ ln1_b,
                           const float* __restrict__ w1, const float* __restrict__ b1,
                           const float* __restrict__ w2, const float* __restrict__ b2,
                           const float* __restrict__ ln2_g, const float* __restrict__ ln2_b,
                           float* __restrict__ out) {
  const int w = threadIdx.x >> 5, lane = threadIdx.x & 31;
  const int row = blockIdx.x * 8 + w;
  if (row >= B_ * Q_) return;
  const int b = row / Q_, qi = row % Q_;

  float av[4];
#pragma unroll
  for (int t = 0; t < 4; ++t) av[t] = att[((size_t)(b * QP_ + qi)) * HD_ + lane * 4 + t];

  float z[4] = {0.f, 0.f, 0.f, 0.f};
  for (int i0 = 0; i0 < HD_; i0 += 4) {
    const int src = i0 >> 2;
    float c0 = __shfl(av[0], src, 32), c1 = __shfl(av[1], src, 32);
    float c2 = __shfl(av[2], src, 32), c3 = __shfl(av[3], src, 32);
#pragma unroll
    for (int t = 0; t < 4; ++t) {
      int j = lane * 4 + t;
      z[t] = fmaf(c0, wp[(i0 + 0) * D_ + j], z[t]);
      z[t] = fmaf(c1, wp[(i0 + 1) * D_ + j], z[t]);
      z[t] = fmaf(c2, wp[(i0 + 2) * D_ + j], z[t]);
      z[t] = fmaf(c3, wp[(i0 + 3) * D_ + j], z[t]);
    }
  }
#pragma unroll
  for (int t = 0; t < 4; ++t) {
    int j = lane * 4 + t;
    z[t] += bp[j] + skip[((size_t)(b * D_ + j)) * Q_ + qi];
  }
  {
    float s = z[0] + z[1] + z[2] + z[3];
    float ss = z[0] * z[0] + z[1] * z[1] + z[2] * z[2] + z[3] * z[3];
    s = redSum32(s); ss = redSum32(ss);
    float mu = s * (1.f / 128.f);
    float rs = rsqrtf(ss * (1.f / 128.f) - mu * mu + 1e-5f);
#pragma unroll
    for (int t = 0; t < 4; ++t) {
      int j = lane * 4 + t;
      z[t] = (z[t] - mu) * rs * ln1_g[j] + ln1_b[j];
    }
  }
  float h[8];
#pragma unroll
  for (int t = 0; t < 8; ++t) h[t] = 0.f;
  for (int i0 = 0; i0 < HD_; i0 += 4) {
    const int src = i0 >> 2;
    float c0 = __shfl(z[0], src, 32), c1 = __shfl(z[1], src, 32);
    float c2 = __shfl(z[2], src, 32), c3 = __shfl(z[3], src, 32);
#pragma unroll
    for (int t = 0; t < 8; ++t) {
      int j2 = lane * 8 + t;
      h[t] = fmaf(c0, w1[(i0 + 0) * 256 + j2], h[t]);
      h[t] = fmaf(c1, w1[(i0 + 1) * 256 + j2], h[t]);
      h[t] = fmaf(c2, w1[(i0 + 2) * 256 + j2], h[t]);
      h[t] = fmaf(c3, w1[(i0 + 3) * 256 + j2], h[t]);
    }
  }
#pragma unroll
  for (int t = 0; t < 8; ++t) {
    float x = h[t] + b1[lane * 8 + t];
    h[t] = 0.5f * x * (1.0f + erff(x * 0.70710678118654752f));
  }
  float y[4] = {z[0], z[1], z[2], z[3]};
  for (int i0 = 0; i0 < 256; i0 += 8) {
    const int src = i0 >> 3;
    float c[8];
#pragma unroll
    for (int u = 0; u < 8; ++u) c[u] = __shfl(h[u], src, 32);
#pragma unroll
    for (int t = 0; t < 4; ++t) {
      int j = lane * 4 + t;
#pragma unroll
      for (int u = 0; u < 8; ++u) y[t] = fmaf(c[u], w2[(i0 + u) * D_ + j], y[t]);
    }
  }
#pragma unroll
  for (int t = 0; t < 4; ++t) y[t] += b2[lane * 4 + t];
  {
    float s = y[0] + y[1] + y[2] + y[3];
    float ss = y[0] * y[0] + y[1] * y[1] + y[2] * y[2] + y[3] * y[3];
    s = redSum32(s); ss = redSum32(ss);
    float mu = s * (1.f / 128.f);
    float rs = rsqrtf(ss * (1.f / 128.f) - mu * mu + 1e-5f);
#pragma unroll
    for (int t = 0; t < 4; ++t) {
      int j = lane * 4 + t;
      out[((size_t)(b * D_ + j)) * Q_ + qi] = (y[t] - mu) * rs * ln2_g[j] + ln2_b[j];
    }
  }
}

extern "C" void kernel_launch(void* const* d_in, const int* in_sizes, int n_in,
                              void* d_out, int out_size, void* d_ws, size_t ws_size,
                              hipStream_t stream) {
  (void)in_sizes; (void)n_in; (void)out_size; (void)ws_size;
  const float* qin   = (const float*)d_in[0];
  const float* kin   = (const float*)d_in[1];
  const float* vin   = (const float*)d_in[2];
  const float* skip  = (const float*)d_in[3];
  const float* lnq_g = (const float*)d_in[4];
  const float* lnq_b = (const float*)d_in[5];
  const float* wq    = (const float*)d_in[6];
  const float* bq    = (const float*)d_in[7];
  const float* lnk_g = (const float*)d_in[8];
  const float* lnk_b = (const float*)d_in[9];
  const float* wk    = (const float*)d_in[10];
  const float* bk    = (const float*)d_in[11];
  const float* lnv_g = (const float*)d_in[12];
  const float* lnv_b = (const float*)d_in[13];
  const float* wv    = (const float*)d_in[14];
  const float* bv    = (const float*)d_in[15];
  const float* wp    = (const float*)d_in[16];
  const float* bp    = (const float*)d_in[17];
  const float* ln1_g = (const float*)d_in[18];
  const float* ln1_b = (const float*)d_in[19];
  const float* w1    = (const float*)d_in[20];
  const float* b1    = (const float*)d_in[21];
  const float* w2    = (const float*)d_in[22];
  const float* b2    = (const float*)d_in[23];
  const float* ln2_g = (const float*)d_in[24];
  const float* ln2_b = (const float*)d_in[25];
  float* out = (float*)d_out;

  char* ws = (char*)d_ws;
  const size_t SZ_WT   = (size_t)D_ * HD_ * 2;                  /* 32768   */
  const size_t SZ_QSUM = (size_t)B_ * QP_ * HD_ * 2;            /* 1286144 */
  const size_t SZ_KP   = (size_t)B_ * M_ * KP_ * DH_ * 2;       /* 1294336 */
  _Float16* wqT  = (_Float16*)(ws);
  _Float16* wkT  = (_Float16*)(ws + SZ_WT);
  _Float16* wvT  = (_Float16*)(ws + 2 * SZ_WT);
  _Float16* qsum = (_Float16*)(ws + 3 * SZ_WT);
  _Float16* kp   = (_Float16*)(ws + 3 * SZ_WT + SZ_QSUM);
  _Float16* vpt  = (_Float16*)(ws + 3 * SZ_WT + SZ_QSUM + SZ_KP);
  float*    att  = (float*)   (ws + 3 * SZ_WT + SZ_QSUM + 2 * SZ_KP);

  k_convert_w<<<(D_ * HD_ + 255) / 256, 256, 0, stream>>>(wq, wk, wv, wqT, wkT, wvT);
  k_proj_q<<<(B_ * QT_ + 3) / 4, 128, 0, stream>>>(qin, lnq_g, lnq_b, wqT, bq, qsum);
  k_proj_kv<<<(B_ * KT_ + 3) / 4, 128, 0, stream>>>(kin, vin, lnk_g, lnk_b, lnv_g, lnv_b,
                                                    wkT, bk, wvT, bv, kp, vpt);
  k_attn<<<(B_ * M_ * QT_ + 7) / 8, 256, 0, stream>>>(qsum, kp, vpt, att);
  k_epilogue<<<(B_ * Q_ + 7) / 8, 256, 0, stream>>>(att, skip, wp, bp, ln1_g, ln1_b,
                                                    w1, b1, w2, b2, ln2_g, ln2_b, out);
}
